// SpatialGATv2_58317065945941
// MI455X (gfx1250) — matrix operand order repaired and checked by reading, not compiled, not run on hardware
//
#include <hip/hip_runtime.h>
#include <hip/hip_bf16.h>
#include <cstdint>
#include <cstddef>

// ---------------------------------------------------------------------------
// GATv2 (3 layers) for MI455X / gfx1250.
// Dense transforms: bf16 WMMA (v_wmma_f32_16x16x32_bf16), wave computes a
// 16x64 tile (A fragment reused across 4 WMMAs per K-step).
// Edge softmax/aggregation: L2-resident float4 gathers + f32 global atomics.
// ---------------------------------------------------------------------------

typedef __attribute__((ext_vector_type(16))) __bf16        v16bf;
typedef __attribute__((ext_vector_type(8)))  float         v8f;
typedef __attribute__((ext_vector_type(8)))  unsigned int  v8u;

#define DEVINL __device__ __forceinline__

static constexpr int NT_C = 10000;   // total nodes (4*2500), divisible by 16
static constexpr int E_C  = 120000;  // edges
static constexpr int FMAX = 512;     // max feature width

DEVINL unsigned int f32_to_bf16_rne(float f) {
  unsigned int u = __float_as_uint(f);
  unsigned int r = u + 0x7FFFu + ((u >> 16) & 1u);
  if ((u & 0x7F800000u) == 0x7F800000u) r = u;  // keep inf/nan pattern
  return r >> 16;
}

DEVINL void atomicMaxF(float* addr, float val) {
  // monotone-in-bits trick: valid with init = -inf
  if (val >= 0.0f) atomicMax((int*)addr, __float_as_int(val));
  else             atomicMin((unsigned int*)addr, __float_as_uint(val));
}

// ---------------- conversion kernels ----------------

// 4 floats -> 4 packed bf16 (two dwords) per thread
__global__ void k_cvt_bf16_v4(const float4* __restrict__ src,
                              uint2* __restrict__ dst, int n4) {
  int i = blockIdx.x * blockDim.x + threadIdx.x;
  if (i >= n4) return;
  float4 v = src[i];
  uint2 o;
  o.x = f32_to_bf16_rne(v.x) | (f32_to_bf16_rne(v.y) << 16);
  o.y = f32_to_bf16_rne(v.z) | (f32_to_bf16_rne(v.w) << 16);
  dst[i] = o;
}

// W (K,N) f32 row-major  ->  Wt (N,K) bf16 row-major
__global__ void k_cvt_bf16_T(const float* __restrict__ W,
                             unsigned short* __restrict__ Wt, int K, int N) {
  int i = blockIdx.x * blockDim.x + threadIdx.x;
  if (i >= K * N) return;
  int k = i / N, n = i - k * N;
  Wt[(size_t)n * K + k] = (unsigned short)f32_to_bf16_rne(W[i]);
}

__global__ void k_fill(float* __restrict__ p, float v, int n) {
  int i = blockIdx.x * blockDim.x + threadIdx.x;
  if (i < n) p[i] = v;
}

// ---------------- WMMA GEMM: C(M,N) = A(M,K) @ Bt(N,K)^T + bias(N) ----------
// One wave computes a 16x64 tile (4 accumulators); A fragment reused across
// the 4 WMMAs of each K-step. Fragment layouts per CDNA5 ISA 7.12.2:
//   A  (bf16 16x32): lane L: M = L&15, half = L>>4; elems 0..7 -> K in
//       [half*8, half*8+8), elems 8..15 -> K in [16+half*8, 16+half*8+8)
//   B  (bf16 32x16): lane L: N = L&15, same K mapping as A (half = L>>4)
//   C/D(f32 16x16):  lane L: N = L&15; elem j -> M = j + 8*(L>>4)

DEVINL v16bf load_bfrag(const unsigned short* p, int half) {
  uint4 b0 = *(const uint4*)(p + half * 8);
  uint4 b1 = *(const uint4*)(p + 16 + half * 8);
  v8u br;
  br[0] = b0.x; br[1] = b0.y; br[2] = b0.z; br[3] = b0.w;
  br[4] = b1.x; br[5] = b1.y; br[6] = b1.z; br[7] = b1.w;
  return __builtin_bit_cast(v16bf, br);
}

__global__ void __launch_bounds__(32)
k_gemm_wmma(const unsigned short* __restrict__ A,    // (M,K) bf16
            const unsigned short* __restrict__ Bt,   // (N,K) bf16
            const float* __restrict__ bias,          // (N)
            float* __restrict__ C,                   // (M,N) f32
            int M, int N, int K) {
  const int lane = threadIdx.x & 31;
  const int tm   = blockIdx.x;
  const int tn0  = blockIdx.y * 4;          // 4 consecutive 16-wide N tiles
  const int loc  = lane & 15;
  const int half = lane >> 4;

  const unsigned short* arow  = A  + (size_t)(tm * 16 + loc) * K;
  const unsigned short* brow0 = Bt + (size_t)(tn0 * 16 + loc) * K;

  v8f acc[4];
#pragma unroll
  for (int j = 0; j < 4; ++j) {
    float b = bias[(tn0 + j) * 16 + loc];
    acc[j] = (v8f){ b, b, b, b, b, b, b, b };
  }

  for (int k0 = 0; k0 < K; k0 += 32) {
    // A fragment: two contiguous 16B chunks, reused for all 4 subtiles
    uint4 a0 = *(const uint4*)(arow + k0 + half * 8);
    uint4 a1 = *(const uint4*)(arow + k0 + 16 + half * 8);
    v8u ar;
    ar[0] = a0.x; ar[1] = a0.y; ar[2] = a0.z; ar[3] = a0.w;
    ar[4] = a1.x; ar[5] = a1.y; ar[6] = a1.z; ar[7] = a1.w;
    v16bf af = __builtin_bit_cast(v16bf, ar);

#pragma unroll
    for (int j = 0; j < 4; ++j) {
      v16bf bf = load_bfrag(brow0 + (size_t)j * 16 * K + k0, half);
      acc[j] = __builtin_amdgcn_wmma_f32_16x16x32_bf16(
          false, af, false, bf, (short)0, acc[j], false, false);
    }
  }

  const int mbase = tm * 16 + half * 8;
#pragma unroll
  for (int j = 0; j < 4; ++j) {
    float* crow = C + (size_t)mbase * N + (tn0 + j) * 16 + loc;
#pragma unroll
    for (int r = 0; r < 8; ++r) crow[(size_t)r * N] = acc[j][r];
  }
}

// ---------------- edge-stage kernels ----------------

// logits[e,h] = sum_c leaky_relu(xl[src,h,c]+xr[dst,h,c]+ew[e]*We[h,c]) * att[h,c]
// + atomic segment-max into m[dst,h].  float4-vectorized over c.
__global__ void k_edge_logits(const float* __restrict__ xl,
                              const float* __restrict__ xr,
                              const int* __restrict__ src,
                              const int* __restrict__ dst,
                              const float* __restrict__ ew,
                              const float* __restrict__ We,
                              const float* __restrict__ att,
                              float* __restrict__ logits,
                              float* __restrict__ mmax,
                              int E, int H, int C) {
  int t = blockIdx.x * blockDim.x + threadIdx.x;
  if (t >= E * H) return;
  int e = t / H, h = t - e * H;
  int F = H * C;
  int s = src[e], d = dst[e];
  float w = ew[e];
  const float4* pl = (const float4*)(xl + (size_t)s * F + h * C);
  const float4* pr = (const float4*)(xr + (size_t)d * F + h * C);
  const float4* pe = (const float4*)(We + h * C);
  const float4* pa = (const float4*)(att + h * C);
  float acc = 0.0f;
  const int C4 = C >> 2;
#pragma unroll 4
  for (int c = 0; c < C4; ++c) {
    float4 l = pl[c], r = pr[c], ee = pe[c], a = pa[c];
    float v0 = l.x + r.x + w * ee.x;
    float v1 = l.y + r.y + w * ee.y;
    float v2 = l.z + r.z + w * ee.z;
    float v3 = l.w + r.w + w * ee.w;
    v0 = (v0 > 0.0f) ? v0 : 0.2f * v0;
    v1 = (v1 > 0.0f) ? v1 : 0.2f * v1;
    v2 = (v2 > 0.0f) ? v2 : 0.2f * v2;
    v3 = (v3 > 0.0f) ? v3 : 0.2f * v3;
    acc = fmaf(v0, a.x, acc);
    acc = fmaf(v1, a.y, acc);
    acc = fmaf(v2, a.z, acc);
    acc = fmaf(v3, a.w, acc);
  }
  logits[t] = acc;
  atomicMaxF(mmax + (size_t)d * H + h, acc);
}

// p = exp(logit - m[dst]); store p over logits; atomic segment-sum into s[dst]
__global__ void k_edge_exp(float* __restrict__ logits,
                           const int* __restrict__ dst,
                           const float* __restrict__ mmax,
                           float* __restrict__ sden,
                           int E, int H) {
  int t = blockIdx.x * blockDim.x + threadIdx.x;
  if (t >= E * H) return;
  int e = t / H, h = t - e * H;
  int d = dst[e];
  float p = __expf(logits[t] - mmax[(size_t)d * H + h]);
  logits[t] = p;
  atomicAdd(sden + (size_t)d * H + h, p);
}

// accum[dst,f..f+3] += xl[src,f..f+3] * p[e,h] / (s[dst,h] + 1e-16)
__global__ void k_edge_scatter(const float* __restrict__ xl,
                               const int* __restrict__ src,
                               const int* __restrict__ dst,
                               const float* __restrict__ p,
                               const float* __restrict__ sden,
                               float* __restrict__ accum,
                               int E, int H, int C) {
  const int F = H * C;
  const int F4 = F >> 2;
  int t = blockIdx.x * blockDim.x + threadIdx.x;
  if (t >= E * F4) return;
  int e = t / F4, q = t - e * F4;
  int f = q << 2;
  int h = f / C;
  int d = dst[e], s = src[e];
  float alpha = p[(size_t)e * H + h] / (sden[(size_t)d * H + h] + 1e-16f);
  float4 v = *(const float4*)(xl + (size_t)s * F + f);
  float* ab = accum + (size_t)d * F + f;
  atomicAdd(ab + 0, v.x * alpha);
  atomicAdd(ab + 1, v.y * alpha);
  atomicAdd(ab + 2, v.z * alpha);
  atomicAdd(ab + 3, v.w * alpha);
}

// out = accum + bias, optional ELU
__global__ void k_finalize(const float* __restrict__ accum,
                           const float* __restrict__ bias,
                           float* __restrict__ out,
                           int total, int F, int do_elu) {
  int t = blockIdx.x * blockDim.x + threadIdx.x;
  if (t >= total) return;
  int f = t % F;
  float v = accum[t] + bias[f];
  if (do_elu) v = (v > 0.0f) ? v : expm1f(v);
  out[t] = v;
}

// ---------------- host driver ----------------

static inline size_t al256(size_t x) { return (x + 255) & ~(size_t)255; }

extern "C" void kernel_launch(void* const* d_in, const int* in_sizes, int n_in,
                              void* d_out, int out_size, void* d_ws, size_t ws_size,
                              hipStream_t stream) {
  (void)in_sizes; (void)n_in; (void)out_size; (void)ws_size;

  const float* x    = (const float*)d_in[0];
  const int*   ei   = (const int*)d_in[1];
  const float* ew   = (const float*)d_in[2];
  const int*   srcI = ei;
  const int*   dstI = ei + E_C;

  // workspace carve-up
  char* wsb = (char*)d_ws;
  size_t o = 0;
  unsigned short* xbf  = (unsigned short*)(wsb + o); o += al256((size_t)NT_C * FMAX * 2);
  unsigned short* wlbf = (unsigned short*)(wsb + o); o += al256((size_t)FMAX * FMAX * 2);
  unsigned short* wrbf = (unsigned short*)(wsb + o); o += al256((size_t)FMAX * FMAX * 2);
  float* xl     = (float*)(wsb + o); o += al256((size_t)NT_C * FMAX * 4);
  float* xr     = (float*)(wsb + o); o += al256((size_t)NT_C * FMAX * 4);  // reused as accum
  float* h1     = (float*)(wsb + o); o += al256((size_t)NT_C * FMAX * 4);
  float* h2     = (float*)(wsb + o); o += al256((size_t)NT_C * FMAX * 4);
  float* logits = (float*)(wsb + o); o += al256((size_t)E_C * 4 * 4);
  float* mmax   = (float*)(wsb + o); o += al256((size_t)NT_C * 4 * 4);
  float* sden   = (float*)(wsb + o); o += al256((size_t)NT_C * 4 * 4);

  const int Kin[3] = { 64, 512, 512 };
  const int Hh[3]  = { 4, 4, 1 };
  const int Cc[3]  = { 128, 128, 256 };

  const float* feat_in[3]  = { x, h1, h2 };
  float*       feat_out[3] = { h1, h2, (float*)d_out };

  const int thr = 256;
  const float NEG_INF = -__builtin_huge_valf();

  for (int l = 0; l < 3; ++l) {
    const int K = Kin[l], H = Hh[l], C = Cc[l], F = H * C;
    const int base = 3 + 7 * l;
    const float* Wl   = (const float*)d_in[base + 0];
    const float* bl   = (const float*)d_in[base + 1];
    const float* Wr   = (const float*)d_in[base + 2];
    const float* br   = (const float*)d_in[base + 3];
    const float* We   = (const float*)d_in[base + 4];
    const float* att  = (const float*)d_in[base + 5];
    const float* bias = (const float*)d_in[base + 6];

    // 1) convert features + weights to bf16 (weights transposed to (N,K))
    const int nA4 = (NT_C * K) >> 2;
    k_cvt_bf16_v4<<<(nA4 + thr - 1) / thr, thr, 0, stream>>>(
        (const float4*)feat_in[l], (uint2*)xbf, nA4);
    const int nW = K * F;
    k_cvt_bf16_T<<<(nW + thr - 1) / thr, thr, 0, stream>>>(Wl, wlbf, K, F);
    k_cvt_bf16_T<<<(nW + thr - 1) / thr, thr, 0, stream>>>(Wr, wrbf, K, F);

    // 2) WMMA GEMMs: xl = x@Wl + bl ; xr = x@Wr + br   (16x64 tile per wave)
    dim3 gg(NT_C / 16, F / 64);
    k_gemm_wmma<<<gg, 32, 0, stream>>>(xbf, wlbf, bl, xl, NT_C, F, K);
    k_gemm_wmma<<<gg, 32, 0, stream>>>(xbf, wrbf, br, xr, NT_C, F, K);

    // 3) segment softmax over incoming edges
    const int nMH = NT_C * H;
    k_fill<<<(nMH + thr - 1) / thr, thr, 0, stream>>>(mmax, NEG_INF, nMH);
    k_fill<<<(nMH + thr - 1) / thr, thr, 0, stream>>>(sden, 0.0f, nMH);

    const int nEH = E_C * H;
    k_edge_logits<<<(nEH + thr - 1) / thr, thr, 0, stream>>>(
        xl, xr, srcI, dstI, ew, We, att, logits, mmax, E_C, H, C);
    k_edge_exp<<<(nEH + thr - 1) / thr, thr, 0, stream>>>(
        logits, dstI, mmax, sden, E_C, H);

    // 4) weighted scatter-aggregate (xr buffer reused as accumulator)
    const int nNF = NT_C * F;
    k_fill<<<(nNF + thr - 1) / thr, thr, 0, stream>>>(xr, 0.0f, nNF);
    const int nEF4 = E_C * (F >> 2);
    k_edge_scatter<<<(nEF4 + thr - 1) / thr, thr, 0, stream>>>(
        xl, srcI, dstI, logits, sden, xr, E_C, H, C);

    // 5) bias + ELU (except last layer)
    k_finalize<<<(nNF + thr - 1) / thr, thr, 0, stream>>>(
        xr, bias, feat_out[l], nNF, F, (l < 2) ? 1 : 0);
  }
}
